// BinaryTreeShInvariantConv_28243704938901
// MI455X (gfx1250) — matrix operand order repaired and checked by reading, not compiled, not run on hardware
//
#include <hip/hip_runtime.h>
#include <hip/hip_bf16.h>

// ---------------------------------------------------------------------------
// SH-invariant conv for MI455X (gfx1250, wave32, WMMA, bf16 datapath).
// Stage-1 GEMM per point (transposed): y^T[32x64] = convK^T[32x32] x patches[32x64]
//   -> D tiles carry SH orders n in each lane's VGPRs => segment reduce is
//      register adds + one ds_swizzle (SWAPX16), no LDS round trip.
// Stage-2 GEMM per 16 pts: out[128x16] = W[128x512] x yl[512x16]
// Both on V_WMMA_F32_16X16X32_BF16 (f32 accumulate). Stage-1 A/B and stage-2 B operands live
// in LDS in *register-fragment order* ([tile][lane][e]); operand fill is two
// ds_load_b128 straight into the 8 operand VGPRs.
// ---------------------------------------------------------------------------

typedef __bf16 bf_t;
typedef bf_t  v16bf __attribute__((ext_vector_type(16)));
typedef float v8f   __attribute__((ext_vector_type(8)));

union Frag {
    unsigned short u[16];
    unsigned       p[8];
    uint4          q[2];
    v16bf          v;
};

enum : int {
    Bb = 8, Vv = 4096, Pp = 32, CIN = 64, NSH = 16, RR = 2, COUT = 128,
    RN = RR * NSH,          // 32
    TV = 16,                // points per workgroup == WMMA N
    KTOT = CIN * RR * 4     // 512 = stage-2 K
};

// truncating f32 -> bf16 (cheap; |rel err| < 2^-7, fine for this op)
__device__ __forceinline__ unsigned short f2bf(float f) {
    return (unsigned short)(__float_as_uint(f) >> 16);
}
// pack two f32 into bf16x2 in one dword: (a>>16) | (b & 0xffff0000)
__device__ __forceinline__ unsigned pk2(float a, float b) {
    return (__float_as_uint(a) >> 16) | (__float_as_uint(b) & 0xffff0000u);
}
// exchange with lane^16 : ds_swizzle SWAPX16 (group-of-32, xor=0x10, and=0x1f)
__device__ __forceinline__ float swap16(float x) {
    return __int_as_float(__builtin_amdgcn_ds_swizzle(__float_as_int(x), 0x401f));
}

__global__ __launch_bounds__(256, 1)
void sh_conv_kernel(const float* __restrict__ signal,
                    const int*   __restrict__ patches_idx,
                    const float* __restrict__ convK,
                    const float* __restrict__ W,
                    const float* __restrict__ bias,
                    float*       __restrict__ out)
{
    extern __shared__ __align__(16) char smem[];
    // LDS (fragment order, [tile][lane(32)][e(16)] halfwords):
    //   Pfrag: 16pt*4ct tiles = 32768 hw (64 KB)   stage-1 B (patches)
    //   Kfrag: 16pt*2r  tiles = 16384 hw (32 KB)   stage-1 A (convK^T)
    //   Yfrag: 16ks     tiles =  8192 hw (16 KB)   stage-2 B (yl)
    unsigned short* Pfrag = (unsigned short*)smem;
    unsigned short* Kfrag = Pfrag + TV * 4 * 32 * 16;
    unsigned short* Yfrag = Kfrag + TV * 2 * 32 * 16;

    const int tid  = threadIdx.x;
    const int wave = tid >> 5;
    const int lane = tid & 31;
    const int col  = lane & 15;     // fragment column
    const int half = lane >> 4;     // lane group

    const int b  = blockIdx.x / (Vv / TV);
    const int v0 = (blockIdx.x % (Vv / TV)) * TV;

    // L2 warm for the weight panel this wave streams in phase 3
    __builtin_prefetch(W + (size_t)(wave * 16) * KTOT, 0, 3);

    // ---- Phase 1a: convK -> stage-1 A fragments (convK^T, bf16) -----------
    // A tile (pt, r): row m = n_sh, K = p.  item = (pt, rn, halfk);
    // consecutive tids take consecutive rn => element loads coalesce.
    {
        const float* ckb = convK + (size_t)(b * Vv + v0) * (Pp * RN);
        for (int j = 0; j < 4; ++j) {
            const int it = tid + 256 * j;
            const int rn = it & 31, halfk = (it >> 5) & 1, pt = it >> 6;
            const int r = rn >> 4, nsh = rn & 15;
            Frag f;
            #pragma unroll
            for (int e = 0; e < 16; ++e) {
                const int p = halfk * 16 + e;
                f.u[e] = f2bf(ckb[(pt * Pp + p) * RN + rn]);
            }
            uint4* dst = (uint4*)(Kfrag + (((pt * 2 + r) * 32 + (nsh + 16 * halfk)) << 4));
            dst[0] = f.q[0]; dst[1] = f.q[1];
        }
    }
    // ---- Phase 1b: gathered patches -> stage-1 B fragments ----------------
    // B tile (pt, ct): K = p (rows), N = c (cols): lane = (c&15)+16*(p>>4),
    // e = p&15.  item = (pt, c, phi); consecutive tids take consecutive c
    // => each row-gather load coalesces across lanes (same row, c+lane).
    {
        const float* sb = signal + (size_t)b * Vv * CIN;
        for (int j = 0; j < 8; ++j) {
            const int it = tid + 256 * j;
            const int c = it & 63, phi = (it >> 6) & 1, pt = it >> 7;
            const int* ib = patches_idx + (((size_t)(b * Vv + v0 + pt)) * Pp) * 2;
            Frag f;
            #pragma unroll
            for (int e = 0; e < 16; ++e) {
                const int p  = phi * 16 + e;
                const int nb = ib[p * 2 + 1];
                f.u[e] = f2bf(sb[(size_t)nb * CIN + c]);
            }
            uint4* dst = (uint4*)(Pfrag + (((pt * 4 + (c >> 4)) * 32 + ((c & 15) + 16 * phi)) << 4));
            dst[0] = f.q[0]; dst[1] = f.q[1];
        }
    }
    __syncthreads();

    // ---- Phase 2: stage-1 WMMA (transposed) + in-register SH reduce -------
    #pragma unroll
    for (int pi = 0; pi < 2; ++pi) {
        const int pt = wave * 2 + pi;
        Frag KF[2];                      // convK^T tiles, reused across ct
        #pragma unroll
        for (int r = 0; r < RR; ++r) {
            const uint4* s = (const uint4*)(Kfrag + (((pt * 2 + r) * 32 + lane) << 4));
            KF[r].q[0] = s[0]; KF[r].q[1] = s[1];
        }
        #pragma unroll
        for (int ct = 0; ct < 4; ++ct) {
            Frag PB;
            {
                const uint4* s = (const uint4*)(Pfrag + (((pt * 4 + ct) * 32 + lane) << 4));
                PB.q[0] = s[0]; PB.q[1] = s[1];
            }
            #pragma unroll
            for (int r = 0; r < RR; ++r) {
                v8f c0 = {};
                v8f d = __builtin_amdgcn_wmma_f32_16x16x32_bf16(
                            false, KF[r].v, false, PB.v, (short)0, c0, false, false);
                // D tile: lane col = channel c, VGPR q = SH order n = q+8*half.
                // half0 holds n0-7, half1 holds n8-15.  Degree sums:
                //   l=0: n0          l=1: n1-3      (half0-local)
                //   l=2: n4-7 + n8   l=3: n9-15     (half1 after one swap)
                const float sq0 = d[0] * d[0];
                const float pB  = d[1] * d[1] + d[2] * d[2] + d[3] * d[3];
                const float pC  = d[4] * d[4] + d[5] * d[5] + d[6] * d[6] + d[7] * d[7];
                const float xC  = swap16(pC);             // other half's pC
                float u0 = half ? (xC + sq0) : sq0;       // s2 | s0
                float u1 = half ? (pB + pC)  : pB;        // s3 | s1
                u0 = __builtin_amdgcn_sqrtf(fmaxf(u0, 1e-4f));
                u1 = __builtin_amdgcn_sqrtf(fmaxf(u1, 1e-4f));
                // k = c*8 + r*4 + l ; half0 -> l=0,1 ; half1 -> l=2,3
                const int c  = ct * 16 + col;
                const int k0 = c * 8 + r * 4;
        *(unsigned*)(Yfrag + ((k0 >> 5) * 32 + (pt + 16 * ((k0 >> 3) & 1))) * 16
                                    + ((k0 >> 4) & 1) * 8 + (k0 & 7) + 2 * half) = pk2(u0, u1);
            }
        }
    }
    __syncthreads();

    // ---- Phase 3: stage-2 WMMA, K=512 in 16 bf16 steps --------------------
    const int mt = wave;                  // 8 waves == 8 Cout tiles
    v8f acc = {};
    for (int ks = 0; ks < KTOT / 32; ++ks) {
        // A = W tile: row i = mt*16+col, K(e) = ks*32 + (e>=8)*16 + half*8 + (e&7)
        const float* wrow = W + (size_t)(mt * 16 + col) * KTOT + ks * 32 + half * 8;
        const float4 f0 = ((const float4*)wrow)[0];
        const float4 f1 = ((const float4*)wrow)[1];
        const float4 f2 = ((const float4*)(wrow + 16))[0];
        const float4 f3 = ((const float4*)(wrow + 16))[1];
        Frag aA;
        aA.p[0] = pk2(f0.x, f0.y); aA.p[1] = pk2(f0.z, f0.w);
        aA.p[2] = pk2(f1.x, f1.y); aA.p[3] = pk2(f1.z, f1.w);
        aA.p[4] = pk2(f2.x, f2.y); aA.p[5] = pk2(f2.z, f2.w);
        aA.p[6] = pk2(f3.x, f3.y); aA.p[7] = pk2(f3.z, f3.w);
        Frag bB;
        {
            const uint4* s = (const uint4*)(Yfrag + ((ks * 32 + lane) << 4));
            bB.q[0] = s[0]; bB.q[1] = s[1];
        }
        acc = __builtin_amdgcn_wmma_f32_16x16x32_bf16(
                  false, aA.v, false, bB.v, (short)0, acc, false, false);
    }
    // bias + ReLU + store: D element (m = q + 8*half, n = col = point)
    #pragma unroll
    for (int q = 0; q < 8; ++q) {
        const int i = mt * 16 + q + half * 8;
        float o = acc[q] + bias[i];
        o = fmaxf(o, 0.f);
        out[((size_t)(b * Vv + v0 + col)) * COUT + i] = o;
    }
}

extern "C" void kernel_launch(void* const* d_in, const int* in_sizes, int n_in,
                              void* d_out, int out_size, void* d_ws, size_t ws_size,
                              hipStream_t stream) {
    const float* signal = (const float*)d_in[0];
    const int*   pidx   = (const int*)d_in[1];
    const float* convK  = (const float*)d_in[2];
    const float* W      = (const float*)d_in[3];
    const float* bias   = (const float*)d_in[4];
    float*       out    = (float*)d_out;

    const size_t shmem =
        (size_t)(TV * 4 * 32 * 16 + TV * 2 * 32 * 16 + 16 * 32 * 16) * sizeof(unsigned short);
                                                                // 114688 B
    (void)hipFuncSetAttribute((const void*)sh_conv_kernel,
                              hipFuncAttributeMaxDynamicSharedMemorySize,
                              (int)shmem);

    const int grid = (Bb * Vv) / TV;                            // 2048
    sh_conv_kernel<<<grid, 256, shmem, stream>>>(signal, pidx, convK, W, bias, out);
}
